// ScaledDotProduct_57062935495339
// MI455X (gfx1250) — matrix operand order repaired and checked by reading, not compiled, not run on hardware
//
#include <hip/hip_runtime.h>
#include <math.h>

typedef __attribute__((ext_vector_type(16))) __bf16        v16bf;
typedef __attribute__((ext_vector_type(8)))  float         v8f;
typedef __attribute__((ext_vector_type(8)))  unsigned int  v8u;

static constexpr int BATCH = 4;
static constexpr int SEQ   = 2048;
static constexpr int DM    = 1024;   // d_model == inp_size
static constexpr int LS    = 40;     // padded LDS row stride (bf16): 80B = 20 banks, conflict-free
static constexpr int BM = 128, BN = 128, BK = 32;
static constexpr int NTHREADS = 256; // 8 waves (wave32)

// ---------------- CDNA5 async global->LDS path (guarded, with fallback) ----------------
#if __has_builtin(__builtin_amdgcn_global_load_async_to_lds_b128) && \
    __has_builtin(__builtin_amdgcn_s_wait_asynccnt)
#define USE_ASYNC 1
typedef int v4i_gcc __attribute__((vector_size(16)));
typedef __attribute__((address_space(1))) v4i_gcc as1_v4i;
typedef __attribute__((address_space(3))) v4i_gcc as3_v4i;
__device__ __forceinline__ void async_cp_b128(const void* g, void* l) {
  __builtin_amdgcn_global_load_async_to_lds_b128((as1_v4i*)g, (as3_v4i*)l, 0, 0);
}
__device__ __forceinline__ void wait_stage() { __builtin_amdgcn_s_wait_asynccnt(0); }
#else
#define USE_ASYNC 0
__device__ __forceinline__ void wait_stage() {}
#endif

// ---------------- helpers ----------------

__device__ __forceinline__ unsigned short f2bf(float f) {
  unsigned int u = __float_as_uint(f);
  u += 0x7fffu + ((u >> 16) & 1u);      // round-to-nearest-even
  return (unsigned short)(u >> 16);
}
__device__ __forceinline__ unsigned int pack2(float a, float b) {
  return (unsigned int)f2bf(a) | ((unsigned int)f2bf(b) << 16);
}

__device__ __forceinline__ v8f wmma_bf16(v16bf a, v16bf b, v8f c) {
  return __builtin_amdgcn_wmma_f32_16x16x32_bf16(false, a, false, b, (short)0, c,
                                                 false, false);
}

// A-fragment (16x32 bf16, ISA layout): lane&15 = M, lane>=16 shifts K by 8.
__device__ __forceinline__ v16bf load_frag_a(const unsigned short* As, int mBase, int lane) {
  const int m = lane & 15, hi = lane >> 4;
  const unsigned short* p = As + (mBase + m) * LS;
  v8u r;
#pragma unroll
  for (int j = 0; j < 4; ++j) r[j]     = *(const unsigned int*)(p + 8 * hi + 2 * j);
#pragma unroll
  for (int j = 0; j < 4; ++j) r[4 + j] = *(const unsigned int*)(p + 16 + 8 * hi + 2 * j);
  return __builtin_bit_cast(v16bf, r);
}

// B-fragment (32x16 bf16): lane&15 = N; lanes>=16 shift K by 8 (same as A). Bs[n][k] k-contiguous.
__device__ __forceinline__ v16bf load_frag_b(const unsigned short* Bs, int nBase, int lane) {
  const int n = lane & 15, hi = lane >> 4;
  const unsigned short* p = Bs + (nBase + n) * LS + 8 * hi;
  v8u r;
#pragma unroll
  for (int j = 0; j < 8; ++j) r[j] = *(const unsigned int*)(p + (j >> 2) * 16 + 2 * (j & 3));
  return __builtin_bit_cast(v16bf, r);
}

#define ZERO_ACC(acc)                                             \
  _Pragma("unroll") for (int i_ = 0; i_ < 2; ++i_)                \
  _Pragma("unroll") for (int j_ = 0; j_ < 4; ++j_)                \
      acc[i_][j_] = (v8f){0.f, 0.f, 0.f, 0.f, 0.f, 0.f, 0.f, 0.f};

#define MMA_STEP(As, Bs, acc, wm, wn, lane)                       \
  {                                                               \
    v16bf afrag_[2], bfrag_[4];                                   \
    _Pragma("unroll") for (int i_ = 0; i_ < 2; ++i_)              \
        afrag_[i_] = load_frag_a(As, (wm) + 16 * i_, lane);       \
    _Pragma("unroll") for (int j_ = 0; j_ < 4; ++j_)              \
        bfrag_[j_] = load_frag_b(Bs, (wn) + 16 * j_, lane);       \
    _Pragma("unroll") for (int i_ = 0; i_ < 2; ++i_)              \
    _Pragma("unroll") for (int j_ = 0; j_ < 4; ++j_)              \
        acc[i_][j_] = wmma_bf16(afrag_[i_], bfrag_[j_], acc[i_][j_]); \
  }

// Stage a 128x32 bf16 tile of row-major data (rowStride elems) into LDS.
// Async path: 2 x b128 per thread straight into LDS, no VGPR round-trip.
__device__ __forceinline__ void stage_rows(const unsigned short* __restrict__ g,
                                           size_t rowStride,
                                           unsigned short* __restrict__ lds, int tid) {
#pragma unroll
  for (int i = 0; i < 2; ++i) {
    const int f = tid + i * NTHREADS;          // 512 x b128
    const int r = f >> 2, c = (f & 3) << 3;
#if USE_ASYNC
    async_cp_b128(g + (size_t)r * rowStride + c, lds + r * LS + c);
#else
    const uint4 v4 = *(const uint4*)(g + (size_t)r * rowStride + c);
    unsigned short* d = lds + r * LS + c;
    *(uint2*)(d)     = make_uint2(v4.x, v4.y);
    *(uint2*)(d + 4) = make_uint2(v4.z, v4.w);
#endif
  }
}

// ---------------- kernel 1: QKV projection (fp32 in -> bf16 out, +bias) ----------------
__device__ __forceinline__ void proj_load(const float* __restrict__ A,
                                          const float* __restrict__ W,
                                          int m0, int n0, int k0, int K, int N,
                                          int tid, float4 ra[4], float4 rw[4]) {
#pragma unroll
  for (int i = 0; i < 4; ++i) {
    const int f = tid + i * NTHREADS;
    const int r = f >> 3, c = (f & 7) << 2;
    ra[i] = *(const float4*)(A + (size_t)(m0 + r) * K + (k0 + c));
    const int kr = f >> 5, nc = (f & 31) << 2;
    rw[i] = *(const float4*)(W + (size_t)(k0 + kr) * N + (n0 + nc));
  }
}
__device__ __forceinline__ void proj_store(unsigned short* __restrict__ As,
                                           unsigned short* __restrict__ Bs,
                                           int tid, const float4 ra[4], const float4 rw[4]) {
#pragma unroll
  for (int i = 0; i < 4; ++i) {
    const int f = tid + i * NTHREADS;
    const int r = f >> 3, c = (f & 7) << 2;
    unsigned short* d = &As[r * LS + c];
    *(unsigned int*)(d)     = pack2(ra[i].x, ra[i].y);
    *(unsigned int*)(d + 2) = pack2(ra[i].z, ra[i].w);
    const int kr = f >> 5, nc = (f & 31) << 2;   // W transposed -> Bs[n][k]
    Bs[(nc + 0) * LS + kr] = f2bf(rw[i].x);
    Bs[(nc + 1) * LS + kr] = f2bf(rw[i].y);
    Bs[(nc + 2) * LS + kr] = f2bf(rw[i].z);
    Bs[(nc + 3) * LS + kr] = f2bf(rw[i].w);
  }
}

__global__ __launch_bounds__(NTHREADS)
void qkv_proj_kernel(const float* __restrict__ A, const float* __restrict__ W,
                     const float* __restrict__ bias, unsigned short* __restrict__ C,
                     int M, int N, int K) {
  __shared__ unsigned short As[2][BM * LS];
  __shared__ unsigned short Bs[2][BN * LS];
  const int tid = threadIdx.x, lane = tid & 31, wid = tid >> 5;
  const int m0 = blockIdx.x * BM, n0 = blockIdx.y * BN;
  const int wm = (wid & 3) * 32, wn = (wid >> 2) * 64;
  v8f acc[2][4];
  ZERO_ACC(acc);

  float4 ra[4], rw[4];
  proj_load(A, W, m0, n0, 0, K, N, tid, ra, rw);
  proj_store(As[0], Bs[0], tid, ra, rw);
  __syncthreads();

  int buf = 0;
  for (int k0 = 0; k0 < K; k0 += BK) {
    const bool more = (k0 + BK) < K;
    if (more) proj_load(A, W, m0, n0, k0 + BK, K, N, tid, ra, rw);  // overlap w/ MMA
    MMA_STEP(As[buf], Bs[buf], acc, wm, wn, lane);
    if (more) proj_store(As[buf ^ 1], Bs[buf ^ 1], tid, ra, rw);    // buf^1 readers passed prior barrier
    __syncthreads();
    buf ^= 1;
  }

  const int hi = lane >> 4, nn = lane & 15;
#pragma unroll
  for (int i = 0; i < 2; ++i)
#pragma unroll
    for (int j = 0; j < 4; ++j) {
      const int col = n0 + wn + 16 * j + nn;
      const float bv = bias[col];
      const int mrow = m0 + wm + 16 * i + 8 * hi;
#pragma unroll
      for (int r = 0; r < 8; ++r)
        C[(size_t)(mrow + r) * N + col] = f2bf(acc[i][j][r] + bv);
    }
}

// ---------------- kernel 2: logits = qp . kp^T * scale ----------------
__global__ __launch_bounds__(NTHREADS)
void logits_kernel(const unsigned short* __restrict__ qp,
                   const unsigned short* __restrict__ kp,
                   float* __restrict__ logits) {
  __shared__ unsigned short As[2][BM * LS];
  __shared__ unsigned short Bs[2][BN * LS];
  const int b = blockIdx.z;
  const unsigned short* Aq = qp + (size_t)b * SEQ * DM;
  const unsigned short* Bk = kp + (size_t)b * SEQ * DM;
  float* Lb = logits + (size_t)b * SEQ * SEQ;
  const int tid = threadIdx.x, lane = tid & 31, wid = tid >> 5;
  const int m0 = blockIdx.x * BM, n0 = blockIdx.y * BN;
  const int wm = (wid & 3) * 32, wn = (wid >> 2) * 64;
  v8f acc[2][4];
  ZERO_ACC(acc);

  const unsigned short* Abase = Aq + (size_t)m0 * DM;
  const unsigned short* Bbase = Bk + (size_t)n0 * DM;  // kp rows == transposed-B layout
  stage_rows(Abase, DM, As[0], tid);
  stage_rows(Bbase, DM, Bs[0], tid);
  wait_stage();
  __syncthreads();

  int buf = 0;
  for (int k0 = 0; k0 < DM; k0 += BK) {
    const bool more = (k0 + BK) < DM;
    if (more) {                                   // async prefetch into buf^1
      stage_rows(Abase + (k0 + BK), DM, As[buf ^ 1], tid);
      stage_rows(Bbase + (k0 + BK), DM, Bs[buf ^ 1], tid);
    }
    MMA_STEP(As[buf], Bs[buf], acc, wm, wn, lane);
    wait_stage();
    __syncthreads();
    buf ^= 1;
  }

  const int hi = lane >> 4, nn = lane & 15;
  const float scale = 0.03125f;   // 1/sqrt(1024)
#pragma unroll
  for (int i = 0; i < 2; ++i)
#pragma unroll
    for (int j = 0; j < 4; ++j) {
      const int col = n0 + wn + 16 * j + nn;
      const int mrow = m0 + wm + 16 * i + 8 * hi;
#pragma unroll
      for (int r = 0; r < 8; ++r)
        Lb[(size_t)(mrow + r) * SEQ + col] = acc[i][j][r] * scale;
    }
}

// ---------------- kernel 3: row softmax (adds mask), fp32 -> bf16 P ----------------
__global__ __launch_bounds__(NTHREADS)
void softmax_kernel(const float* __restrict__ logits, const float* __restrict__ mask,
                    unsigned short* __restrict__ P) {
  const int row = blockIdx.x;             // b*SEQ + q
  const int q = row & (SEQ - 1);
  const float* x = logits + (size_t)row * SEQ;
  const float* mr = mask + (size_t)q * SEQ;   // mask [1,S,S] broadcasts over batch
  unsigned short* pr = P + (size_t)row * SEQ;
  const int tid = threadIdx.x, lane = tid & 31, wid = tid >> 5;

  float v[8];
  float mx = -3.4e38f;
#pragma unroll
  for (int i = 0; i < 8; ++i) {
    const int idx = tid + i * NTHREADS;
    v[i] = x[idx] + mr[idx];
    mx = fmaxf(mx, v[i]);
  }
#pragma unroll
  for (int off = 16; off > 0; off >>= 1) mx = fmaxf(mx, __shfl_xor(mx, off, 32));
  __shared__ float redm[8], reds[8];
  if (lane == 0) redm[wid] = mx;
  __syncthreads();
  mx = redm[0];
#pragma unroll
  for (int w = 1; w < 8; ++w) mx = fmaxf(mx, redm[w]);

  float s = 0.f;
#pragma unroll
  for (int i = 0; i < 8; ++i) { v[i] = __expf(v[i] - mx); s += v[i]; }
#pragma unroll
  for (int off = 16; off > 0; off >>= 1) s += __shfl_xor(s, off, 32);
  if (lane == 0) reds[wid] = s;
  __syncthreads();
  float tot = 0.f;
#pragma unroll
  for (int w = 0; w < 8; ++w) tot += reds[w];
  const float inv = 1.0f / tot;
#pragma unroll
  for (int i = 0; i < 8; ++i) pr[tid + i * NTHREADS] = f2bf(v[i] * inv);
}

// ---------------- kernel 4: out = P . vp (fp32 output) ----------------
__device__ __forceinline__ void av_loadB(const unsigned short* __restrict__ Bv,
                                         int n0, int k0, int tid, uint4 rb[2]) {
#pragma unroll
  for (int i = 0; i < 2; ++i) {
    const int f = tid + i * NTHREADS;
    const int kr = f >> 4, nc = (f & 15) << 3;
    rb[i] = *(const uint4*)(Bv + (size_t)(k0 + kr) * DM + n0 + nc);
  }
}
__device__ __forceinline__ void av_storeB(unsigned short* __restrict__ Bs,
                                          int tid, const uint4 rb[2]) {
#pragma unroll
  for (int i = 0; i < 2; ++i) {
    const int f = tid + i * NTHREADS;
    const int kr = f >> 4, nc = (f & 15) << 3;
    const unsigned int w4[4] = {rb[i].x, rb[i].y, rb[i].z, rb[i].w};
#pragma unroll
    for (int t = 0; t < 4; ++t) {   // transpose vp[k][n] -> Bs[n][k]
      Bs[(nc + 2 * t) * LS + kr]     = (unsigned short)(w4[t] & 0xffffu);
      Bs[(nc + 2 * t + 1) * LS + kr] = (unsigned short)(w4[t] >> 16);
    }
  }
}

__global__ __launch_bounds__(NTHREADS)
void av_kernel(const unsigned short* __restrict__ P,
               const unsigned short* __restrict__ vp,
               float* __restrict__ out) {
  __shared__ unsigned short As[2][BM * LS];
  __shared__ unsigned short Bs[2][BN * LS];
  const int b = blockIdx.z;
  const unsigned short* Ap = P + (size_t)b * SEQ * SEQ;
  const unsigned short* Bv = vp + (size_t)b * SEQ * DM;
  float* Ob = out + (size_t)b * SEQ * DM;
  const int tid = threadIdx.x, lane = tid & 31, wid = tid >> 5;
  const int m0 = blockIdx.x * BM, n0 = blockIdx.y * BN;
  const int wm = (wid & 3) * 32, wn = (wid >> 2) * 64;
  v8f acc[2][4];
  ZERO_ACC(acc);

  const unsigned short* Abase = Ap + (size_t)m0 * SEQ;
  uint4 rb[2];
  stage_rows(Abase, SEQ, As[0], tid);            // P rows: async-able verbatim copy
  av_loadB(Bv, n0, 0, tid, rb);
  av_storeB(Bs[0], tid, rb);
  wait_stage();
  __syncthreads();

  int buf = 0;
  for (int k0 = 0; k0 < SEQ; k0 += BK) {
    const bool more = (k0 + BK) < SEQ;
    if (more) {
      stage_rows(Abase + (k0 + BK), SEQ, As[buf ^ 1], tid);
      av_loadB(Bv, n0, k0 + BK, tid, rb);
    }
    MMA_STEP(As[buf], Bs[buf], acc, wm, wn, lane);
    if (more) av_storeB(Bs[buf ^ 1], tid, rb);
    wait_stage();
    __syncthreads();
    buf ^= 1;
  }

  const int hi = lane >> 4, nn = lane & 15;
#pragma unroll
  for (int i = 0; i < 2; ++i)
#pragma unroll
    for (int j = 0; j < 4; ++j) {
      const int col = n0 + wn + 16 * j + nn;
      const int mrow = m0 + wm + 16 * i + 8 * hi;
#pragma unroll
      for (int r = 0; r < 8; ++r)
        Ob[(size_t)(mrow + r) * DM + col] = acc[i][j][r];
    }
}

// ---------------- launch ----------------
extern "C" void kernel_launch(void* const* d_in, const int* in_sizes, int n_in,
                              void* d_out, int out_size, void* d_ws, size_t ws_size,
                              hipStream_t stream) {
  const float* q        = (const float*)d_in[0];
  const float* k        = (const float*)d_in[1];
  const float* v        = (const float*)d_in[2];
  const float* mask     = (const float*)d_in[3];
  const float* kernel_q = (const float*)d_in[4];
  const float* bias_q   = (const float*)d_in[5];
  const float* kernel_k = (const float*)d_in[6];
  const float* bias_k   = (const float*)d_in[7];
  const float* kernel_v = (const float*)d_in[8];
  const float* bias_v   = (const float*)d_in[9];

  // workspace: [0,16M) qp bf16 | [16M,32M) kp bf16 | [32M,48M) vp bf16 |
  //            [48M,112M) logits fp32 ; P bf16 (32MB) reuses [0,32M).
  const size_t projElems = (size_t)BATCH * SEQ * DM;   // 8,388,608
  unsigned short* qp = (unsigned short*)d_ws;
  unsigned short* kp = qp + projElems;
  unsigned short* vp = kp + projElems;
  float* logits = (float*)((char*)d_ws + 3 * projElems * sizeof(unsigned short));
  unsigned short* P = (unsigned short*)d_ws;

  const dim3 blk(NTHREADS);
  const int M = BATCH * SEQ;   // 8192

  const dim3 gproj(M / BM, DM / BN, 1);                 // (64, 8)
  qkv_proj_kernel<<<gproj, blk, 0, stream>>>(q, kernel_q, bias_q, qp, M, DM, DM);
  qkv_proj_kernel<<<gproj, blk, 0, stream>>>(k, kernel_k, bias_k, kp, M, DM, DM);
  qkv_proj_kernel<<<gproj, blk, 0, stream>>>(v, kernel_v, bias_v, vp, M, DM, DM);

  const dim3 glog(SEQ / BM, SEQ / BN, BATCH);           // (16, 16, 4)
  logits_kernel<<<glog, blk, 0, stream>>>(qp, kp, logits);

  softmax_kernel<<<dim3(BATCH * SEQ), blk, 0, stream>>>(logits, mask, P);

  const dim3 gav(SEQ / BM, DM / BN, BATCH);             // (16, 8, 4)
  av_kernel<<<gav, blk, 0, stream>>>(P, vp, (float*)d_out);
}
